// GlobalGraph_63110249447604
// MI455X (gfx1250) — compile-verified, matrix operand order checked by reading, not run on hardware
//
#include <hip/hip_runtime.h>
#include <math.h>

typedef __attribute__((ext_vector_type(16))) _Float16 v16h;
typedef __attribute__((ext_vector_type(8)))  float    v8f;
typedef __attribute__((ext_vector_type(4)))  float    v4f;

#define C_DIM 128

#define WMMA_F16(a, b, c) \
  __builtin_amdgcn_wmma_f32_16x16x32_f16(false, (a), false, (b), (short)0, (c), false, false)

union V16HU { v16h v; v4f f[2]; };

// Load one 16x32 f16 operand tile in the documented CDNA5 16-bit A/B VGPR layout:
// lane L<16  -> row L,    K offsets {0..7, 16..23}
// lane L>=16 -> row L-16, K offsets {8..15, 24..31}
// `base` points at (row0, k0) of a row-major f16 array with leading dim `ld` (halves).
__device__ __forceinline__ v16h load_tile_h(const _Float16* base, int ld, int lane) {
  const _Float16* p = base + (size_t)(lane & 15) * ld + ((lane & 16) ? 8 : 0);
  V16HU r;
  r.f[0] = *(const v4f*)p;          // halves 0..7  (16B aligned)
  r.f[1] = *(const v4f*)(p + 16);   // halves 8..15
  return r.v;
}

// ---------------------------------------------------------------------------
// Kernel 0a: zero the final output (harness poisons d_out).
// ---------------------------------------------------------------------------
__global__ void zero_kernel(float* __restrict__ p, long long n) {
  long long i = (long long)blockIdx.x * blockDim.x + threadIdx.x;
  if (i < n) p[i] = 0.0f;
}

// ---------------------------------------------------------------------------
// Kernel 0b: one-time f32 -> f16 conversion (x and the three weight mats).
// Each thread converts 8 contiguous elements: 2x b128 load, 1x b128 store.
// ---------------------------------------------------------------------------
__global__ void cvt_f16_kernel(const float* __restrict__ src,
                               _Float16* __restrict__ dst, long long n8) {
  const long long i = (long long)blockIdx.x * blockDim.x + threadIdx.x;
  if (i >= n8) return;
  const v4f a = *(const v4f*)(src + i * 8);
  const v4f b = *(const v4f*)(src + i * 8 + 4);
  v16h h;  // only low 8 halves used
#pragma unroll
  for (int k = 0; k < 4; ++k) { h[k] = (_Float16)a[k]; h[4 + k] = (_Float16)b[k]; }
  *(v4f*)(dst + i * 8) = *(v4f*)&h;  // 16 bytes = 8 halves
}

// ---------------------------------------------------------------------------
// Kernel 1: QKV projection.  q = x @ Wq^T + bq  (likewise k, v).
// One wave per 16x64 output strip (4 WMMA tiles share each A load).
// blockIdx.y: z = y>>1 (0=Q,1=K,2=V), col group = y&1 (64 cols each).
// Q, K stored row-major f16 [N][C]; V stored transposed f16 [B][C][T].
// ---------------------------------------------------------------------------
__global__ __launch_bounds__(32)
void qkv_proj_kernel(const _Float16* __restrict__ xh,
                     const _Float16* __restrict__ Wh,   // [3][C][C] f16
                     const float* __restrict__ bq, const float* __restrict__ bk,
                     const float* __restrict__ bv,
                     _Float16* __restrict__ Qh, _Float16* __restrict__ Kh,
                     _Float16* __restrict__ Vt, int T) {
  const int lane = threadIdx.x;
  const int row0 = blockIdx.x * 16;            // node rows
  const int z    = blockIdx.y >> 1;            // 0=Q 1=K 2=V
  const int col0 = (blockIdx.y & 1) * 64;      // output channel strip
  const _Float16* W = Wh + (size_t)z * C_DIM * C_DIM;
  const float* bias = (z == 0) ? bq : (z == 1) ? bk : bv;

  v8f acc[4];
#pragma unroll
  for (int c = 0; c < 4; ++c)
#pragma unroll
    for (int i = 0; i < 8; ++i) acc[c][i] = 0.0f;

  // D(16m x 16n) = A(x: 16m x 32k) * B(W^T: 32k x 16n); B col n = W row (col0+n).
#pragma unroll
  for (int k0 = 0; k0 < C_DIM; k0 += 32) {
    const v16h a = load_tile_h(xh + (size_t)row0 * C_DIM + k0, C_DIM, lane);
#pragma unroll
    for (int c = 0; c < 4; ++c) {
      const v16h b = load_tile_h(W + (size_t)(col0 + c * 16) * C_DIM + k0, C_DIM, lane);
      acc[c] = WMMA_F16(a, b, acc[c]);
    }
  }

  const int col = lane & 15;                   // N index (C/D layout)
  const int hi  = lane >> 4;                   // M offset 0/8
#pragma unroll
  for (int c = 0; c < 4; ++c) {
    const int cc = col0 + c * 16 + col;        // global output channel
    const float bb = bias[cc];
#pragma unroll
    for (int v = 0; v < 8; ++v) {
      const int row = row0 + v + 8 * hi;       // global node index
      const float val = acc[c][v] + bb;
      if (z == 0) {
        Qh[(size_t)row * C_DIM + cc] = (_Float16)val;
      } else if (z == 1) {
        Kh[(size_t)row * C_DIM + cc] = (_Float16)val;
      } else {
        const int bIdx = row / T, t = row % T; // transpose V per batch
        Vt[((size_t)bIdx * C_DIM + cc) * T + t] = (_Float16)val;
      }
    }
  }
}

// ---------------------------------------------------------------------------
// Kernel 2: masked softmax attention, flash style.
// One wave = 16 query rows x full C=128 output, online softmax over s.
// ---------------------------------------------------------------------------
__global__ __launch_bounds__(32)
void attention_kernel(const _Float16* __restrict__ Qh,
                      const _Float16* __restrict__ Kh,
                      const _Float16* __restrict__ Vt,
                      const int* __restrict__ valid_lens,
                      float* __restrict__ outA, int T) {
  const int lane = threadIdx.x;
  const int b    = blockIdx.y;
  const int t0   = blockIdx.x * 16;
  const int vl   = valid_lens[b];
  const int col  = lane & 15;
  const int hi   = lane >> 4;

  __shared__ _Float16 lds[16 * 32];            // P repack buffer (C/D -> A layout)

  // Q tile: 16 rows x 128 cols as four 16x32 A operands.
  v16h qa[4];
#pragma unroll
  for (int kk = 0; kk < 4; ++kk)
    qa[kk] = load_tile_h(Qh + (size_t)(b * T + t0) * C_DIM + kk * 32, C_DIM, lane);

  v8f acc[8];
#pragma unroll
  for (int c = 0; c < 8; ++c)
#pragma unroll
    for (int i = 0; i < 8; ++i) acc[c][i] = 0.0f;

  float m[8], l[8];
#pragma unroll
  for (int v = 0; v < 8; ++v) { m[v] = -INFINITY; l[v] = 0.0f; }

  for (int s0 = 0; s0 < vl; s0 += 32) {
    // scores: two 16x16 tiles (cols s0..s0+15, s0+16..s0+31)
    v8f sa, sb;
#pragma unroll
    for (int i = 0; i < 8; ++i) { sa[i] = 0.0f; sb[i] = 0.0f; }
#pragma unroll
    for (int kk = 0; kk < 4; ++kk) {
      const v16h kb0 = load_tile_h(Kh + (size_t)(b * T + s0)      * C_DIM + kk * 32, C_DIM, lane);
      const v16h kb1 = load_tile_h(Kh + (size_t)(b * T + s0 + 16) * C_DIM + kk * 32, C_DIM, lane);
      sa = WMMA_F16(qa[kk], kb0, sa);
      sb = WMMA_F16(qa[kk], kb1, sb);
    }
    // mask: reference sets cols >= vl to -1e6 (exp underflows to exactly 0 in f32)
    if (s0 + col >= vl) {
#pragma unroll
      for (int i = 0; i < 8; ++i) sa[i] = -1000000.0f;
    }
    if (s0 + 16 + col >= vl) {
#pragma unroll
      for (int i = 0; i < 8; ++i) sb[i] = -1000000.0f;
    }

    // online softmax per row; row v lives in one 16-lane half -> shfl_xor width 16
    float p0[8], p1[8];
#pragma unroll
    for (int v = 0; v < 8; ++v) {
      float rmax = fmaxf(sa[v], sb[v]);
#pragma unroll
      for (int off = 8; off >= 1; off >>= 1)
        rmax = fmaxf(rmax, __shfl_xor(rmax, off, 16));
      const float mn    = fmaxf(m[v], rmax);
      const float alpha = __expf(m[v] - mn);
      m[v] = mn;
      const float e0 = __expf(sa[v] - mn);
      const float e1 = __expf(sb[v] - mn);
      p0[v] = e0; p1[v] = e1;
      float rs = e0 + e1;
#pragma unroll
      for (int off = 8; off >= 1; off >>= 1)
        rs += __shfl_xor(rs, off, 16);
      l[v] = l[v] * alpha + rs;
#pragma unroll
      for (int c = 0; c < 8; ++c) acc[c][v] *= alpha;   // rescale running output
    }

    // repack P (C/D layout) -> row-major LDS [16][32] -> A layout
#pragma unroll
    for (int v = 0; v < 8; ++v) {
      const int row = v + 8 * hi;
      lds[row * 32 + col]      = (_Float16)p0[v];
      lds[row * 32 + 16 + col] = (_Float16)p1[v];
    }
    __syncthreads();
    const v16h pa = load_tile_h(lds, 32, lane);
    __syncthreads();

    // out += P(16x32 over s) * V(32s x 16c); Vt is [B][C][T] so s is contiguous
#pragma unroll
    for (int c = 0; c < 8; ++c) {
      const v16h vb = load_tile_h(Vt + ((size_t)b * C_DIM + c * 16) * T + s0, T, lane);
      acc[c] = WMMA_F16(pa, vb, acc[c]);
    }
  }

  // normalize and store fp32 attention output
#pragma unroll
  for (int v = 0; v < 8; ++v) {
    const float inv = 1.0f / l[v];
    const int row = b * T + t0 + v + 8 * hi;
#pragma unroll
    for (int c = 0; c < 8; ++c)
      outA[(size_t)row * C_DIM + c * 16 + col] = acc[c][v] * inv;
  }
}

// ---------------------------------------------------------------------------
// Kernel 3: edge scatter-add  out[dst] += attnOut[src].
// One thread per (edge, 4-channel chunk): float4 gather + 4 f32 atomics (L2).
// ---------------------------------------------------------------------------
__global__ void scatter_kernel(const float* __restrict__ outA,
                               const int* __restrict__ src,
                               const int* __restrict__ dst,
                               float* __restrict__ out, long long E) {
  const long long idx = (long long)blockIdx.x * blockDim.x + threadIdx.x;
  const long long e = idx >> 5;            // C/4 = 32 quads per edge
  const int q = (int)(idx & 31);
  if (e >= E) return;
  const int s = src[e];
  const int d = dst[e];
  const v4f val = *(const v4f*)(outA + (size_t)s * C_DIM + q * 4);
  float* po = out + (size_t)d * C_DIM + q * 4;
  atomicAdd(po + 0, val[0]);
  atomicAdd(po + 1, val[1]);
  atomicAdd(po + 2, val[2]);
  atomicAdd(po + 3, val[3]);
}

// ---------------------------------------------------------------------------
extern "C" void kernel_launch(void* const* d_in, const int* in_sizes, int n_in,
                              void* d_out, int out_size, void* d_ws, size_t ws_size,
                              hipStream_t stream) {
  const float* x          = (const float*)d_in[0];
  const int*   edge_index = (const int*)d_in[1];
  const int*   valid_lens = (const int*)d_in[2];
  // d_in[3] = time_step_len scalar on device; shapes derived from sizes instead
  const float* Wq = (const float*)d_in[4];
  const float* bq = (const float*)d_in[5];
  const float* Wk = (const float*)d_in[6];
  const float* bk = (const float*)d_in[7];
  const float* Wv = (const float*)d_in[8];
  const float* bv = (const float*)d_in[9];

  const int C = C_DIM;
  const int N = in_sizes[0] / C;          // 32768
  const int B = in_sizes[2];              // 32
  const int T = N / B;                    // 1024
  const long long E = in_sizes[1] / 2;    // 524288

  // Workspace: xh | Wh | Qh | Kh | Vt (f16) | attention output (f32)
  _Float16* xh = (_Float16*)d_ws;
  _Float16* Wh = xh + (size_t)N * C;                 // [3][C][C]
  _Float16* Qh = Wh + (size_t)3 * C * C;
  _Float16* Kh = Qh + (size_t)N * C;
  _Float16* Vt = Kh + (size_t)N * C;
  float*    outA = (float*)(Vt + (size_t)N * C);
  float*    out  = (float*)d_out;

  // 0a) zero final output (atomic accumulation target)
  {
    const long long n = (long long)N * C;
    zero_kernel<<<(unsigned)((n + 255) / 256), 256, 0, stream>>>(out, n);
  }
  // 0b) one-time f32 -> f16 conversion of x and weights
  {
    const long long nx8 = (long long)N * C / 8;
    cvt_f16_kernel<<<(unsigned)((nx8 + 255) / 256), 256, 0, stream>>>(x, xh, nx8);
    const long long nw8 = (long long)C * C / 8;
    cvt_f16_kernel<<<(unsigned)((nw8 + 255) / 256), 256, 0, stream>>>(Wq, Wh, nw8);
    cvt_f16_kernel<<<(unsigned)((nw8 + 255) / 256), 256, 0, stream>>>(Wk, Wh + (size_t)C * C, nw8);
    cvt_f16_kernel<<<(unsigned)((nw8 + 255) / 256), 256, 0, stream>>>(Wv, Wh + (size_t)2 * C * C, nw8);
  }
  // 1) QKV projection: (N/16) row tiles x (3 matrices * 2 col strips of 64)
  {
    dim3 grid(N / 16, 6);
    qkv_proj_kernel<<<grid, 32, 0, stream>>>(xh, Wh, bq, bk, bv, Qh, Kh, Vt, T);
  }
  // 2) attention: (T/16) query tiles x B batches
  {
    dim3 grid(T / 16, B);
    attention_kernel<<<grid, 32, 0, stream>>>(Qh, Kh, Vt, valid_lens, outA, T);
  }
  // 3) edge scatter-add
  {
    const long long total = E * 32;
    scatter_kernel<<<(unsigned)((total + 255) / 256), 256, 0, stream>>>(
        outA, edge_index, edge_index + E, out, E);
  }
}
